// HolographicMemoryBank_51857435131956
// MI455X (gfx1250) — matrix operand order repaired and checked by reading, not compiled, not run on hardware
//
#include <hip/hip_runtime.h>
#include <hip/hip_bf16.h>
#include <math.h>

typedef __attribute__((ext_vector_type(16))) _Float16 v16h;
typedef __attribute__((ext_vector_type(8)))  _Float16 v8h;
typedef __attribute__((ext_vector_type(8)))  float    v8f;

#define N_MEM  4096
#define N_WAVE 64
#define N_MODE 128
#define BATCH  512

// ---------------------------------------------------------------------------
// K1: decoded[n,m] = sum_w holo[n,w,m]*conj(keys[n,w]);
//     dec_norm_f16[n,m] = |decoded|, normB[n] = ||dec_norm[n,:]||
// One block per memory n, 128 threads (one per mode). Coalesced across m.
// Hologram is 256 MB streamed exactly once -> non-temporal loads so it does
// not evict decN/scores from the 192 MB L2.
// ---------------------------------------------------------------------------
__global__ void decode_kernel(const float* __restrict__ hr,
                              const float* __restrict__ hi,
                              const float* __restrict__ kr,
                              const float* __restrict__ ki,
                              _Float16* __restrict__ decN,
                              float* __restrict__ normB) {
    __shared__ float skr[N_WAVE];
    __shared__ float ski[N_WAVE];
    __shared__ float red[N_MODE];
    const int n = blockIdx.x;
    const int m = threadIdx.x;                 // 0..127
    if (m < N_WAVE) {
        skr[m] = kr[(size_t)n * N_WAVE + m];
        ski[m] = ki[(size_t)n * N_WAVE + m];
    }
    __syncthreads();

    const float* hrp = hr + (size_t)n * N_WAVE * N_MODE + m;
    const float* hip = hi + (size_t)n * N_WAVE * N_MODE + m;
    float are = 0.f, aim = 0.f;
#pragma unroll 4
    for (int w = 0; w < N_WAVE; ++w) {
        const float re = __builtin_nontemporal_load(hrp + (size_t)w * N_MODE);
        const float im = __builtin_nontemporal_load(hip + (size_t)w * N_MODE);
        const float cr = skr[w], ci = ski[w];
        // (re + i*im) * (cr - i*ci)
        are = fmaf(re, cr, fmaf(im,  ci, are));
        aim = fmaf(im, cr, fmaf(-re, ci, aim));
    }
    const float mag = sqrtf(are * are + aim * aim);
    decN[(size_t)n * N_MODE + m] = (_Float16)mag;

    red[m] = mag * mag;
    __syncthreads();
    for (int s = N_MODE / 2; s > 0; s >>= 1) {
        if (m < s) red[m] += red[m + s];
        __syncthreads();
    }
    if (m == 0) normB[n] = sqrtf(red[0]);
}

// ---------------------------------------------------------------------------
// K2: cue_norm_f16[b,m] = |cue[b,m]|, normA[b] = ||cue_norm[b,:]||
// ---------------------------------------------------------------------------
__global__ void cue_norm_kernel(const float* __restrict__ cue,
                                _Float16* __restrict__ cueN,
                                float* __restrict__ normA) {
    __shared__ float red[N_MODE];
    const int b = blockIdx.x;
    const int m = threadIdx.x;
    const float v = fabsf(cue[(size_t)b * N_MODE + m]);
    cueN[(size_t)b * N_MODE + m] = (_Float16)v;
    red[m] = v * v;
    __syncthreads();
    for (int s = N_MODE / 2; s > 0; s >>= 1) {
        if (m < s) red[m] += red[m + s];
        __syncthreads();
    }
    if (m == 0) normA[b] = sqrtf(red[0]);
}

// ---------------------------------------------------------------------------
// K3: scores[b,n] = |sum_m cueN[b,m]*decN[n,m]| / (normA[b]*normB[n] + 1e-10)
// via v_wmma_f32_16x16x32_f16. One wave = one 16x16 tile, K=128 (4 WMMAs).
// A fragment (16x32 f16): lane row = lane&15; halves 0..15 -> K=k0+hi16*16+h
//   (contiguous k order, same as B)
// B fragment (32x16 f16): lane col = lane&15; halves 0..15 -> K=k0+hi16*16+h
// Epilogue uses v_rcp_f32 instead of IEEE divide (score precision is ample).
// ---------------------------------------------------------------------------
__global__ void score_wmma_kernel(const _Float16* __restrict__ cueN,  // [512,128]
                                  const _Float16* __restrict__ decN,  // [4096,128]
                                  const float* __restrict__ normA,
                                  const float* __restrict__ normB,
                                  float* __restrict__ scores) {       // [512,4096]
    const int wid  = blockIdx.x * (blockDim.x >> 5) + (threadIdx.x >> 5);
    const int lane = threadIdx.x & 31;
    const int bt = wid >> 8;            // 0..31   batch tile
    const int nt = wid & 255;           // 0..255  memory tile
    const int b0 = bt * 16, n0 = nt * 16;
    const int lh   = lane & 15;
    const int hi16 = lane >> 4;         // 0 or 1

    const _Float16* arow = cueN + (size_t)(b0 + lh) * N_MODE + hi16 * 16;
    const _Float16* brow = decN + (size_t)(n0 + lh) * N_MODE + hi16 * 16;

    v8f acc = {};
#pragma unroll
    for (int k0 = 0; k0 < N_MODE; k0 += 32) {
        const v8h alo = *(const v8h*)(arow + k0);        // K = k0+hi16*16 .. +7
        const v8h ahi = *(const v8h*)(arow + k0 + 8);    // K = k0+hi16*16+8 .. +15
        const v16h a = __builtin_shufflevector(alo, ahi,
                0, 1, 2, 3, 4, 5, 6, 7, 8, 9, 10, 11, 12, 13, 14, 15);
        const v16h b = *(const v16h*)(brow + k0);        // K = k0+hi16*16 .. +15
        acc = __builtin_amdgcn_wmma_f32_16x16x32_f16(
                /*neg_a=*/false, a, /*neg_b=*/false, b,
                /*c_mod=*/(short)0, acc, /*reuse_a=*/false, /*reuse_b=*/false);
    }

    const int   ncol = n0 + lh;         // C/D: lane -> column N
    const float nb   = normB[ncol];
#pragma unroll
    for (int r = 0; r < 8; ++r) {       // C/D: VGPR r -> row M (+8 for hi lanes)
        const int brw = b0 + r + hi16 * 8;
        const float inv = __builtin_amdgcn_rcpf(fmaf(normA[brw], nb, 1e-10f));
        scores[(size_t)brw * N_MEM + ncol] = fabsf(acc[r]) * inv;
    }
}

// ---------------------------------------------------------------------------
// K4: per-batch max + argmax over 4096 memories (first-max tie break).
// ---------------------------------------------------------------------------
__global__ void argmax_kernel(const float* __restrict__ scores,
                              float* __restrict__ conf,
                              int* __restrict__ bidx) {
    __shared__ float sv[256];
    __shared__ int   si[256];
    const int b = blockIdx.x, t = threadIdx.x;
    float best = -1.f;
    int   bi   = N_MEM;
    for (int n = t; n < N_MEM; n += 256) {
        const float s = scores[(size_t)b * N_MEM + n];
        if (s > best) { best = s; bi = n; }     // increasing n keeps first max
    }
    sv[t] = best; si[t] = bi;
    __syncthreads();
    for (int s = 128; s > 0; s >>= 1) {
        if (t < s) {
            const float ov = sv[t + s]; const int oi = si[t + s];
            if (ov > sv[t] || (ov == sv[t] && oi < si[t])) { sv[t] = ov; si[t] = oi; }
        }
        __syncthreads();
    }
    if (t == 0) { conf[b] = sv[0]; bidx[b] = si[0]; }
}

// ---------------------------------------------------------------------------
// K5: recalled[b,w,m] = holo[best[b],w,m] * conj(keys[best[b],w])
// Output interleaved (re,im) pairs = complex64 flat layout. Output (32 MB)
// is never re-read -> non-temporal stores.
// ---------------------------------------------------------------------------
__global__ void recall_kernel(const float* __restrict__ hr,
                              const float* __restrict__ hi,
                              const float* __restrict__ kr,
                              const float* __restrict__ ki,
                              const int* __restrict__ bidx,
                              float* __restrict__ out) {
    const size_t g = (size_t)blockIdx.x * blockDim.x + threadIdx.x; // 0 .. 512*8192-1
    const int b = (int)(g >> 13);
    const int r = (int)(g & 8191);
    const int w = r >> 7;
    const int m = r & 127;
    const int idx = bidx[b];
    const size_t src = ((size_t)idx * N_WAVE + w) * N_MODE + m;
    const float re = hr[src], im = hi[src];
    const float cr = kr[(size_t)idx * N_WAVE + w];
    const float ci = ki[(size_t)idx * N_WAVE + w];
    __builtin_nontemporal_store(fmaf(re, cr,  im * ci), out + 2 * g);     // Re
    __builtin_nontemporal_store(fmaf(im, cr, -re * ci), out + 2 * g + 1); // Im
}

// ---------------------------------------------------------------------------
extern "C" void kernel_launch(void* const* d_in, const int* in_sizes, int n_in,
                              void* d_out, int out_size, void* d_ws, size_t ws_size,
                              hipStream_t stream) {
    (void)in_sizes; (void)n_in; (void)out_size; (void)ws_size;

    const float* hr  = (const float*)d_in[0];   // [4096,64,128]
    const float* hi  = (const float*)d_in[1];   // [4096,64,128]
    const float* kr  = (const float*)d_in[2];   // [4096,64]
    const float* ki  = (const float*)d_in[3];   // [4096,64]
    const float* cue = (const float*)d_in[4];   // [512,128]

    float* out = (float*)d_out;
    // recalled: 2*512*64*128 floats, then confidence (512 f32), then idx (512 i32)
    float* out_conf = out + (size_t)2 * BATCH * N_WAVE * N_MODE;
    int*   out_idx  = (int*)(out_conf + BATCH);

    char* ws = (char*)d_ws;
    _Float16* decN  = (_Float16*)ws;                                  // 1 MB, 32B aligned
    _Float16* cueN  = (_Float16*)(ws + (size_t)(1 << 20));            // 128 KB
    float*    normB = (float*)(ws + (size_t)(1 << 20) + (128 << 10)); // 16 KB
    float*    normA = normB + N_MEM;                                  // 2 KB
    float*    scores = normA + BATCH;                                 // 8 MB

    decode_kernel  <<<N_MEM, N_MODE, 0, stream>>>(hr, hi, kr, ki, decN, normB);
    cue_norm_kernel<<<BATCH, N_MODE, 0, stream>>>(cue, cueN, normA);
    score_wmma_kernel<<<(32 * 256) / 8, 256, 0, stream>>>(cueN, decN, normA, normB, scores);
    argmax_kernel  <<<BATCH, 256, 0, stream>>>(scores, out_conf, out_idx);
    recall_kernel  <<<(BATCH * N_WAVE * N_MODE) / 256, 256, 0, stream>>>(
            hr, hi, kr, ki, out_idx, out);
}
